// Net_30932354466331
// MI455X (gfx1250) — matrix operand order repaired and checked by reading, not compiled, not run on hardware
//
#include <hip/hip_runtime.h>

// ---------------------------------------------------------------------------
// KAN forward (7 layers) for MI455X / gfx1250, wave32, bf16 WMMA 16x16x32.
// Each layer:  y = silu(x) @ bw.T + einsum('big,oig->bo', bspl(x), sw*sc)
//            = Aaug(B, 9*IN) @ Waug(9*IN, OUT)   (K index = feat*9 + {silu, 8 bases})
// Weights are packed per-launch into WMMA-B-fragment-native 32x16 tiles and
// staged into LDS with double-buffered global_load_async_to_lds_b128 so the
// copy of k-tile t+1 overlaps the WMMAs of k-tile t (ASYNCcnt pipeline).
// ---------------------------------------------------------------------------

typedef __bf16 bf16_t;
typedef __bf16 v16bf __attribute__((ext_vector_type(16)));
typedef __bf16 v8bf  __attribute__((ext_vector_type(8)));
typedef float  v8f   __attribute__((ext_vector_type(8)));

#define KAN_BATCH 32768
#define GH 0.4f              // grid spacing h = 2/GRID_SIZE

#define A_RS     296         // LDS row stride of A stage, halves (288 + pad, conflict-free)
#define A_BYTES  (64 * A_RS * 2)   // 37,888 B

// ---------------------------------------------------------------------------
// Pack augmented weights into WMMA-B-fragment-native tiles:
// tile (nt, kt) is 32(K) x 16(N); lane l holds 16 contiguous halves for
// n = l%16, k in ISA table order (8h+s, 16+8h+s-8), tile order tileIdx = nt*KT + kt.
// ---------------------------------------------------------------------------
__global__ void pack_weights_kernel(const float* __restrict__ bw,
                                    const float* __restrict__ sw,
                                    const float* __restrict__ sc,
                                    bf16_t* __restrict__ W,
                                    int IN, int OUT, int KT, int NT)
{
    const long total = (long)NT * KT * 512;
    for (long idx = blockIdx.x * (long)blockDim.x + threadIdx.x; idx < total;
         idx += (long)gridDim.x * blockDim.x) {
        long tile   = idx >> 9;
        int  within = (int)(idx & 511);
        int  ln = within >> 4, es = within & 15;
        int  kk = ((ln >> 4) << 3) + (es & 7) + ((es >> 3) << 4), nn = ln & 15;
        int  kt = (int)(tile % KT), nt = (int)(tile / KT);
        int  k  = kt * 32 + kk,     n  = nt * 16 + nn;
        float v = 0.0f;
        if (n < OUT) {
            int  i  = k / 9;
            int  q  = k - 9 * i;
            long oi = (long)n * IN + i;
            v = (q == 0) ? bw[oi] : sw[oi * 8 + (q - 1)] * sc[oi];
        }
        W[idx] = (bf16_t)v;
    }
}

// ---------------------------------------------------------------------------
// Async-copy one k-tile's weight block (NT tiles x 1KB) into an LDS buffer.
// Each lane copies 16B segments; tracked by ASYNCcnt.
// ---------------------------------------------------------------------------
template <int NT, int KT>
__device__ __forceinline__ void stage_B_async(const bf16_t* __restrict__ W,
                                              int gkt, unsigned dstBase, int tid)
{
    for (int s = tid; s < NT * 64; s += 256) {
        const int nt = s >> 6, w = s & 63;
        const unsigned ldsoff = dstBase + (unsigned)(s << 4);
        const unsigned long long g = (unsigned long long)(uintptr_t)
            ((const char*)W + (((long)nt * KT + gkt) << 10) + ((long)w << 4));
        asm volatile("global_load_async_to_lds_b128 %0, %1, off"
                     :: "v"(ldsoff), "v"(g) : "memory");
    }
}

__device__ __forceinline__ void wait_async_zero()
{
    asm volatile("s_wait_asynccnt 0" ::: "memory");
}

// ---------------------------------------------------------------------------
// Fused KAN layer.  grid.x = BATCH/64, block = 256 (8 waves: 4 M x 2 N-groups)
// Dynamic LDS: [A stage 37,888 B][B buf0 NT KB][B buf1 NT KB]
// ---------------------------------------------------------------------------
template <int IN, int OUT, int OUTP>
__global__ __launch_bounds__(256)
void kan_fused(const float* __restrict__ X,
               const bf16_t* __restrict__ W,
               float* __restrict__ Y, int ldY)
{
    constexpr int K9      = IN * 9;
    constexpr int KT_TOT  = K9 / 32;       // exact: 9*IN % 32 == 0 for IN in {64,128,256}
    constexpr int NT      = OUTP / 16;
    constexpr int NCH     = IN / 32;
    constexpr int NTB     = NT * 1024;     // bytes per B buffer
    constexpr int NTW_MAX = (NT + 1) / 2;

    extern __shared__ char smem[];
    bf16_t* sA = (bf16_t*)smem;
    const unsigned ldsBase = (unsigned)(uintptr_t)smem;  // flat[31:0] == LDS offset

    const int tid  = threadIdx.x;
    const int lane = tid & 31;
    const int wid  = tid >> 5;
    const int mt   = wid >> 1;             // M-tile 0..3
    const int nh   = wid & 1;              // N-tile parity owned by this wave
    const long rb  = (long)blockIdx.x * 64;

    v8f acc[NTW_MAX];
#pragma unroll
    for (int j = 0; j < NTW_MAX; ++j)
#pragma unroll
        for (int r = 0; r < 8; ++r) acc[j][r] = 0.0f;

    // A-fragment addressing (ISA 16-bit A 16x32 layout):
    //  lanes 0-15 : K 0-7 then 16-23 ; lanes 16-31 : K 8-15 then 24-31
    const int arow = mt * 16 + (lane & 15);
    const int asel = (lane < 16) ? 0 : 8;

    // prologue: start copy of the first weight k-tile block
    stage_B_async<NT, KT_TOT>(W, 0, ldsBase + A_BYTES, tid);

    for (int ch = 0; ch < NCH; ++ch) {
        __syncthreads();   // previous chunk's LDS reads complete
        // ---- stage 64 rows x 32 features -> 64 x 288 bf16 in LDS ----
        for (int p = tid; p < 64 * 32; p += 256) {
            const int f   = p & 31;        // coalesced over features
            const int row = p >> 5;
            const float x = X[(rb + row) * IN + ch * 32 + f];

            const float sil = x / (1.0f + __expf(-x));   // silu

            // cubic B-spline bases on uniform grid g(j) = (j-3)*GH - 1
            float b[11];
#pragma unroll
            for (int j = 0; j < 11; ++j) {
                const float gj  = GH * (float)(j - 3) - 1.0f;
                const float gj1 = GH * (float)(j - 2) - 1.0f;
                b[j] = (x >= gj && x < gj1) ? 1.0f : 0.0f;
            }
#pragma unroll
            for (int d = 1; d <= 3; ++d) {
                const float inv = 1.0f / ((float)d * GH);
#pragma unroll
                for (int t = 0; t + d < 11; ++t) {
                    const float gt   = GH * (float)(t - 3) - 1.0f;
                    const float gtd1 = GH * (float)(t + d - 2) - 1.0f;
                    b[t] = (x - gt) * inv * b[t] + (gtd1 - x) * inv * b[t + 1];
                }
            }
            bf16_t* dst = &sA[row * A_RS + f * 9];
            dst[0] = (bf16_t)sil;
#pragma unroll
            for (int q = 0; q < 8; ++q) dst[1 + q] = (bf16_t)b[q];
        }
        wait_async_zero();   // weight block for first k-tile of this chunk landed
        __syncthreads();     // A staged + B visible to all waves

        // ---- GEMM over this 288-wide K chunk (9 k-tiles of 32) ----
        for (int kt = 0; kt < 9; ++kt) {
            const int gkt = ch * 9 + kt;

            // overlap: copy next k-tile's weights into the other buffer
            if (gkt + 1 < KT_TOT)
                stage_B_async<NT, KT_TOT>(W, gkt + 1,
                                          ldsBase + A_BYTES + ((gkt + 1) & 1) * NTB, tid);

            const int kb = kt * 32;
            const v8bf alo = *(const v8bf*)&sA[arow * A_RS + kb + asel];
            const v8bf ahi = *(const v8bf*)&sA[arow * A_RS + kb + asel + 16];
            const v16bf afrag = __builtin_shufflevector(
                alo, ahi, 0, 1, 2, 3, 4, 5, 6, 7, 8, 9, 10, 11, 12, 13, 14, 15);

            const char* bbuf = smem + A_BYTES + (gkt & 1) * NTB;
#pragma unroll
            for (int j = 0; j < NTW_MAX; ++j) {
                const int nt = nh + 2 * j;
                if (nt < NT) {   // wave-uniform
                    const v16bf bfrag = *(const v16bf*)(bbuf + nt * 1024 + lane * 32);
                    acc[j] = __builtin_amdgcn_wmma_f32_16x16x32_bf16(
                        false, afrag, false, bfrag, (short)0, acc[j], false, false);
                }
            }
            wait_async_zero();   // next buffer's copies (this wave's share) landed
            __syncthreads();     // all shares landed; current buffer free to reuse
        }
    }

    // ---- store 64 x OUT (mask padded columns) ----
#pragma unroll
    for (int j = 0; j < NTW_MAX; ++j) {
        const int nt = nh + 2 * j;
        if (nt < NT) {
            const int n = nt * 16 + (lane & 15);
            if (n < OUT) {
#pragma unroll
                for (int r = 0; r < 8; ++r) {
                    const int m = mt * 16 + r + asel;   // C/D layout: vgpr r, lane half
                    Y[(rb + m) * (long)ldY + n] = acc[j][r];
                }
            }
        }
    }
}

// ---------------------------------------------------------------------------
// Host-side orchestration
// ---------------------------------------------------------------------------
static inline int kan_smem_bytes(int NT) { return A_BYTES + 2 * NT * 1024; }

extern "C" void kernel_launch(void* const* d_in, const int* in_sizes, int n_in,
                              void* d_out, int out_size, void* d_ws, size_t ws_size,
                              hipStream_t stream)
{
    (void)in_sizes; (void)n_in; (void)out_size; (void)ws_size;

    struct LCfg { int IN, OUT, OUTP; };
    const LCfg cfg[7] = { {128, 256, 256}, {256, 256, 256}, {256, 64, 64},
                          {64, 64, 64},   {64, 18, 32},     {64, 64, 64},
                          {64, 1, 16} };

    const float* x = (const float*)d_in[0];
    float* out     = (float*)d_out;

    // workspace layout (all section sizes are multiples of 256 bytes)
    char* ws = (char*)d_ws;
    size_t off = 0;
    bf16_t* W[7];
    for (int l = 0; l < 7; ++l) {
        const int KT = cfg[l].IN * 9 / 32;
        const int NT = cfg[l].OUTP / 16;
        W[l] = (bf16_t*)(ws + off);
        off += (size_t)NT * KT * 512 * sizeof(bf16_t);
    }
    float* h1 = (float*)(ws + off); off += (size_t)KAN_BATCH * 256 * sizeof(float);
    float* h2 = (float*)(ws + off); off += (size_t)KAN_BATCH * 256 * sizeof(float);
    float* h3 = (float*)(ws + off); off += (size_t)KAN_BATCH * 64  * sizeof(float);
    float* t  = (float*)(ws + off); off += (size_t)KAN_BATCH * 64  * sizeof(float);

    // allow >64KB dynamic LDS for the 256-out layers (idempotent, capture-safe)
    (void)hipFuncSetAttribute((const void*)&kan_fused<128, 256, 256>,
                              hipFuncAttributeMaxDynamicSharedMemorySize, kan_smem_bytes(16));
    (void)hipFuncSetAttribute((const void*)&kan_fused<256, 256, 256>,
                              hipFuncAttributeMaxDynamicSharedMemorySize, kan_smem_bytes(16));

    // pack augmented weights (per launch; cheap, ~1.1M elements total)
    for (int l = 0; l < 7; ++l) {
        const int IN = cfg[l].IN, OUT = cfg[l].OUT, OUTP = cfg[l].OUTP;
        const int KT = IN * 9 / 32, NT = OUTP / 16;
        const long total = (long)NT * KT * 512;
        const int grid = (int)((total + 255) / 256);
        pack_weights_kernel<<<grid, 256, 0, stream>>>(
            (const float*)d_in[1 + 3 * l], (const float*)d_in[2 + 3 * l],
            (const float*)d_in[3 + 3 * l], W[l], IN, OUT, KT, NT);
    }

    const int NB = KAN_BATCH / 64;   // 512 workgroups

    // trunk
    kan_fused<128, 256, 256><<<NB, 256, kan_smem_bytes(16), stream>>>(x,  W[0], h1, 256);
    kan_fused<256, 256, 256><<<NB, 256, kan_smem_bytes(16), stream>>>(h1, W[1], h2, 256);
    kan_fused<256, 64,  64 ><<<NB, 256, kan_smem_bytes(4),  stream>>>(h2, W[2], h3, 64);

    // advantage head -> d_out[0 : B*18)
    kan_fused<64, 64, 64><<<NB, 256, kan_smem_bytes(4), stream>>>(h3, W[3], t, 64);
    kan_fused<64, 18, 32><<<NB, 256, kan_smem_bytes(2), stream>>>(t,  W[4], out, 18);

    // state-value head -> d_out[B*18 : B*19)
    kan_fused<64, 64, 64><<<NB, 256, kan_smem_bytes(4), stream>>>(h3, W[5], t, 64);
    kan_fused<64, 1,  16><<<NB, 256, kan_smem_bytes(1), stream>>>(t,  W[6],
                                                                  out + (long)KAN_BATCH * 18, 1);
}
